// TransformerBlock_8177617731635
// MI455X (gfx1250) — compile-verified, matrix operand order checked by reading, not run on hardware
//
#include <hip/hip_runtime.h>
#include <math.h>

#define B_  8
#define N_  2048
#define H_  1024
#define FF_ 4096

typedef __attribute__((ext_vector_type(16))) __bf16 bf16x16;
typedef __attribute__((ext_vector_type(8)))  float  f32x8;

__device__ __forceinline__ f32x8 wmma_bf16(bf16x16 a, bf16x16 b, f32x8 c) {
  // D = A x B + C, 16x16x32 bf16 -> f32 (wave32)
  return __builtin_amdgcn_wmma_f32_16x16x32_bf16(
      false, a, false, b, (short)0, c, false, false);
}

// Async DMA: global -> LDS, 16 bytes per active lane (ASYNCcnt-tracked).
// LDS operand = low 32 bits of the generic shared address (== LDS byte offset).
__device__ __forceinline__ void async_ld_b128(void* lds, const void* gptr) {
  unsigned l = (unsigned)(uintptr_t)lds;
  asm volatile("global_load_async_to_lds_b128 %0, %1, off"
               :: "v"(l), "v"(gptr) : "memory");
}
__device__ __forceinline__ void wait_async0() {
  asm volatile("s_wait_asynccnt 0x0" ::: "memory");
}

// Gather a 16x32 bf16 fragment from LDS (row stride 32 elems).
// Lanes 0-15: row M=lane, K {0..7,16..23}; lanes 16-31: same rows, K {8..15,24..31}.
__device__ __forceinline__ bf16x16 frag_ld32(const __bf16* base, int lane) {
  const int r  = lane & 15;
  const int kb = (lane & 16) ? 8 : 0;
  const __bf16* p = base + r * 32 + kb;
  union { bf16x16 v; uint4 u[2]; } t;
  t.u[0] = *(const uint4*)(p);
  t.u[1] = *(const uint4*)(p + 16);
  return t.v;
}

__device__ __forceinline__ float gelu_exact(float x) {
  return 0.5f * x * (1.0f + erff(x * 0.70710678118654752f));
}

__device__ __forceinline__ f32x8 zero8() {
  f32x8 z = {0.f, 0.f, 0.f, 0.f, 0.f, 0.f, 0.f, 0.f};
  return z;
}

// ---------------------------------------------------------------------------
// Shared GEMM core: C_tile[128x128] += A[m0.., :K] * Bt[n0.., :K]^T
// A row-major bf16 (lda), Bt row-major bf16 (ldb) holding B transposed.
// 256 threads = 8 waves; wave (w&3) -> 32-row strip, (w>>2) -> 64-col strip.
// Double-buffered LDS tiles filled with GLOBAL_LOAD_ASYNC_TO_LDS_B128.
// ---------------------------------------------------------------------------
__device__ __forceinline__ void gemm_tiles(const __bf16* __restrict__ A, long lda,
                                           const __bf16* __restrict__ Bt, long ldb,
                                           int m0, int n0, int K,
                                           f32x8 (&acc)[2][4]) {
  __shared__ __bf16 As[2][128 * 32];
  __shared__ __bf16 Bs[2][128 * 32];
  const int tid  = threadIdx.x;
  const int lane = tid & 31;
  const int wave = tid >> 5;
  const int wr = (wave & 3) * 32;
  const int wc = (wave >> 2) * 64;
  const int row = tid >> 1;          // 0..127
  const int c0  = (tid & 1) * 16;    // 0 or 16 (elems); 32B per thread per tile

  const __bf16* aSrc = A  + (size_t)(m0 + row) * lda + c0;
  const __bf16* bSrc = Bt + (size_t)(n0 + row) * ldb + c0;
  __bf16* aDst0 = &As[0][row * 32 + c0];
  __bf16* bDst0 = &Bs[0][row * 32 + c0];
  __bf16* aDst1 = &As[1][row * 32 + c0];
  __bf16* bDst1 = &Bs[1][row * 32 + c0];

  // prologue: fill buffer 0 for k0 = 0
  async_ld_b128(aDst0,     aSrc);
  async_ld_b128(aDst0 + 8, aSrc + 8);
  async_ld_b128(bDst0,     bSrc);
  async_ld_b128(bDst0 + 8, bSrc + 8);

  int cur = 0;
  for (int k0 = 0; k0 < K; k0 += 32) {
    wait_async0();
    __syncthreads();
    if (k0 + 32 < K) {
      const __bf16* an = aSrc + k0 + 32;
      const __bf16* bn = bSrc + k0 + 32;
      __bf16* ad = cur ? aDst0 : aDst1;
      __bf16* bd = cur ? bDst0 : bDst1;
      async_ld_b128(ad,     an);
      async_ld_b128(ad + 8, an + 8);
      async_ld_b128(bd,     bn);
      async_ld_b128(bd + 8, bn + 8);
    }
    const __bf16* Ab = As[cur];
    const __bf16* Bb = Bs[cur];
    bf16x16 a0 = frag_ld32(Ab + (wr +  0) * 32, lane);
    bf16x16 a1 = frag_ld32(Ab + (wr + 16) * 32, lane);
    for (int j = 0; j < 4; j++) {
      bf16x16 bf = frag_ld32(Bb + (wc + j * 16) * 32, lane);
      acc[0][j] = wmma_bf16(a0, bf, acc[0][j]);
      acc[1][j] = wmma_bf16(a1, bf, acc[1][j]);
    }
    cur ^= 1;
  }
}

// ---------------------------------------------------------------------------
// Prep: elementwise fp32 -> bf16
// ---------------------------------------------------------------------------
__global__ void cvt_bf16_kernel(const float* __restrict__ src, __bf16* __restrict__ dst, long n) {
  long i = (long)blockIdx.x * blockDim.x + threadIdx.x;
  const long stride = (long)gridDim.x * blockDim.x;
  for (; i < n; i += stride) dst[i] = (__bf16)src[i];
}

// ---------------------------------------------------------------------------
// Prep: transpose + convert: dst[c][r] = (bf16)src[r][c], per z-batch.
// grid (C/32, R/32, batches), block 256
// ---------------------------------------------------------------------------
__global__ void transpose_cvt_kernel(const float* __restrict__ src, __bf16* __restrict__ dst,
                                     int R, int C) {
  __shared__ float tile[32][33];
  const size_t zoff = (size_t)blockIdx.z * R * C;
  src += zoff; dst += zoff;
  const int c0 = blockIdx.x * 32, r0 = blockIdx.y * 32;
  const int tx = threadIdx.x & 31, ty = threadIdx.x >> 5;  // ty 0..7
  for (int q = 0; q < 32; q += 8)
    tile[ty + q][tx] = src[(size_t)(r0 + ty + q) * C + c0 + tx];
  __syncthreads();
  for (int q = 0; q < 32; q += 8)
    dst[(size_t)(c0 + ty + q) * R + r0 + tx] = (__bf16)tile[tx][ty + q];
}

// ---------------------------------------------------------------------------
// Kernel: S[b] = x[b] @ x[b]^T   grid (N/128, N/128, B)
// ---------------------------------------------------------------------------
__global__ void scores_kernel(const __bf16* __restrict__ xbf, float* __restrict__ S) {
  const int b  = blockIdx.z;
  const int m0 = blockIdx.y * 128;
  const int n0 = blockIdx.x * 128;
  const __bf16* xb = xbf + (size_t)b * N_ * H_;
  f32x8 acc[2][4];
  for (int i = 0; i < 2; i++) for (int j = 0; j < 4; j++) acc[i][j] = zero8();

  gemm_tiles(xb, H_, xb, H_, m0, n0, H_, acc);

  const int lane = threadIdx.x & 31, wave = threadIdx.x >> 5;
  const int wr = (wave & 3) * 32, wc = (wave >> 2) * 64;
  const int ln = lane & 15, hi = (lane & 16) ? 8 : 0;
  float* Sb = S + (size_t)b * N_ * N_;
  for (int i = 0; i < 2; i++)
    for (int j = 0; j < 4; j++) {
      const int col = n0 + wc + j * 16 + ln;
      for (int r = 0; r < 8; r++)
        Sb[(size_t)(m0 + wr + i * 16 + r + hi) * N_ + col] = acc[i][j][r];
    }
}

// ---------------------------------------------------------------------------
// Kernel: row softmax; P (bf16) written in place over each S row
// (P row stride = 2*N_ bf16 elems == N_ floats). grid B*N, block 256.
// ---------------------------------------------------------------------------
__global__ void softmax_kernel(float* __restrict__ S) {
  __shared__ float sd[256];
  const int row = blockIdx.x;
  float* srow = S + (size_t)row * N_;
  __bf16* prow = (__bf16*)srow;
  const int t = threadIdx.x;

  float v[8];
  float m = -3.402823466e38f;
  for (int q = 0; q < 8; q++) { v[q] = srow[t + 256 * q]; m = fmaxf(m, v[q]); }
  sd[t] = m; __syncthreads();
  for (int s = 128; s > 0; s >>= 1) { if (t < s) sd[t] = fmaxf(sd[t], sd[t + s]); __syncthreads(); }
  m = sd[0]; __syncthreads();

  float sum = 0.f;
  for (int q = 0; q < 8; q++) { v[q] = expf(v[q] - m); sum += v[q]; }
  sd[t] = sum; __syncthreads();
  for (int s = 128; s > 0; s >>= 1) { if (t < s) sd[t] += sd[t + s]; __syncthreads(); }
  const float inv = 1.0f / sd[0];
  __syncthreads();
  for (int q = 0; q < 8; q++) prow[t + 256 * q] = (__bf16)(v[q] * inv);
}

// ---------------------------------------------------------------------------
// Kernel: att[b] = P[b] @ x[b]  (B operand = xT_bf[b][h][n]) grid (H/128, N/128, B)
// ---------------------------------------------------------------------------
__global__ void attend_kernel(const __bf16* __restrict__ P, const __bf16* __restrict__ xT,
                              float* __restrict__ att) {
  const int b  = blockIdx.z;
  const int m0 = blockIdx.y * 128;
  const int n0 = blockIdx.x * 128;
  const __bf16* Ab  = P  + (size_t)b * N_ * 2 * N_;   // row stride 2N bf16
  const __bf16* Btb = xT + (size_t)b * H_ * N_;       // [H][N] row-major
  f32x8 acc[2][4];
  for (int i = 0; i < 2; i++) for (int j = 0; j < 4; j++) acc[i][j] = zero8();

  gemm_tiles(Ab, 2 * N_, Btb, N_, m0, n0, N_, acc);

  const int lane = threadIdx.x & 31, wave = threadIdx.x >> 5;
  const int wr = (wave & 3) * 32, wc = (wave >> 2) * 64;
  const int ln = lane & 15, hi = (lane & 16) ? 8 : 0;
  float* ob = att + (size_t)b * N_ * H_;
  for (int i = 0; i < 2; i++)
    for (int j = 0; j < 4; j++) {
      const int col = n0 + wc + j * 16 + ln;
      for (int r = 0; r < 8; r++)
        ob[(size_t)(m0 + wr + i * 16 + r + hi) * H_ + col] = acc[i][j][r];
    }
}

// ---------------------------------------------------------------------------
// Kernel: out = LayerNorm(a + bres)*g + beta; optionally also bf16 copy.
// grid B*N, block 256
// ---------------------------------------------------------------------------
__global__ void add_ln_kernel(const float* __restrict__ a, const float* __restrict__ bres,
                              const float* __restrict__ gg, const float* __restrict__ bb,
                              float* __restrict__ out, __bf16* __restrict__ out_bf) {
  __shared__ float sd[256];
  const int row = blockIdx.x;
  const int t = threadIdx.x;
  const float* ar = a    + (size_t)row * H_;
  const float* br = bres + (size_t)row * H_;
  float v[4]; float s = 0.f;
  for (int q = 0; q < 4; q++) { const int i = t + 256 * q; v[q] = ar[i] + br[i]; s += v[q]; }
  sd[t] = s; __syncthreads();
  for (int st = 128; st > 0; st >>= 1) { if (t < st) sd[t] += sd[t + st]; __syncthreads(); }
  const float mu = sd[0] * (1.0f / H_); __syncthreads();
  float s2 = 0.f;
  for (int q = 0; q < 4; q++) { const float d = v[q] - mu; s2 += d * d; }
  sd[t] = s2; __syncthreads();
  for (int st = 128; st > 0; st >>= 1) { if (t < st) sd[t] += sd[t + st]; __syncthreads(); }
  const float rstd = rsqrtf(sd[0] * (1.0f / H_) + 1e-5f);
  float* orow = out + (size_t)row * H_;
  for (int q = 0; q < 4; q++) {
    const int i = t + 256 * q;
    const float y = (v[q] - mu) * rstd * gg[i] + bb[i];
    orow[i] = y;
    if (out_bf) out_bf[(size_t)row * H_ + i] = (__bf16)y;
  }
}

// ---------------------------------------------------------------------------
// Kernel: g = gelu(h_bf @ w1t^T + b1) bf16.  grid (FF/128, B*N/128)
// ---------------------------------------------------------------------------
__global__ void ffn1_kernel(const __bf16* __restrict__ hbf, const __bf16* __restrict__ w1t,
                            const float* __restrict__ b1, __bf16* __restrict__ g) {
  const int m0 = blockIdx.y * 128;
  const int n0 = blockIdx.x * 128;
  f32x8 acc[2][4];
  for (int i = 0; i < 2; i++) for (int j = 0; j < 4; j++) acc[i][j] = zero8();

  gemm_tiles(hbf, H_, w1t, H_, m0, n0, H_, acc);

  const int lane = threadIdx.x & 31, wave = threadIdx.x >> 5;
  const int wr = (wave & 3) * 32, wc = (wave >> 2) * 64;
  const int ln = lane & 15, hi = (lane & 16) ? 8 : 0;
  for (int i = 0; i < 2; i++)
    for (int j = 0; j < 4; j++) {
      const int col = n0 + wc + j * 16 + ln;
      const float bias = b1[col];
      for (int r = 0; r < 8; r++)
        g[(size_t)(m0 + wr + i * 16 + r + hi) * FF_ + col] =
            (__bf16)gelu_exact(acc[i][j][r] + bias);
    }
}

// ---------------------------------------------------------------------------
// Kernel: f2 = g @ w2t^T + b2 (fp32).  grid (H/128, B*N/128)
// ---------------------------------------------------------------------------
__global__ void ffn2_kernel(const __bf16* __restrict__ g, const __bf16* __restrict__ w2t,
                            const float* __restrict__ b2, float* __restrict__ f2) {
  const int m0 = blockIdx.y * 128;
  const int n0 = blockIdx.x * 128;
  f32x8 acc[2][4];
  for (int i = 0; i < 2; i++) for (int j = 0; j < 4; j++) acc[i][j] = zero8();

  gemm_tiles(g, FF_, w2t, FF_, m0, n0, FF_, acc);

  const int lane = threadIdx.x & 31, wave = threadIdx.x >> 5;
  const int wr = (wave & 3) * 32, wc = (wave >> 2) * 64;
  const int ln = lane & 15, hi = (lane & 16) ? 8 : 0;
  for (int i = 0; i < 2; i++)
    for (int j = 0; j < 4; j++) {
      const int col = n0 + wc + j * 16 + ln;
      const float bias = b2[col];
      for (int r = 0; r < 8; r++)
        f2[(size_t)(m0 + wr + i * 16 + r + hi) * H_ + col] = acc[i][j][r] + bias;
    }
}

// ---------------------------------------------------------------------------
#define MB(x) ((size_t)(x) << 20)

extern "C" void kernel_launch(void* const* d_in, const int* in_sizes, int n_in,
                              void* d_out, int out_size, void* d_ws, size_t ws_size,
                              hipStream_t stream) {
  const float* x     = (const float*)d_in[0];
  const float* ln1_g = (const float*)d_in[1];
  const float* ln1_b = (const float*)d_in[2];
  const float* ln2_g = (const float*)d_in[3];
  const float* ln2_b = (const float*)d_in[4];
  const float* w1    = (const float*)d_in[5];
  const float* b1    = (const float*)d_in[6];
  const float* w2    = (const float*)d_in[7];
  const float* b2    = (const float*)d_in[8];
  float* out = (float*)d_out;

  char* ws = (char*)d_ws;
  // ws layout (368 MB):
  //   [  0,128): S fp32 [B,N,N]; reused as P bf16 (in place), then g bf16 [B*N,FF]
  //   [128,192): att fp32; reused as f2
  //   [192,256): h fp32
  //   [256,288): x_bf   [B,N,H]
  //   [288,320): xT_bf  [B,H,N]
  //   [320,352): h_bf   [B*N,H]
  //   [352,360): w1t bf16 [FF,H]
  //   [360,368): w2t bf16 [H,FF]
  float*  S    = (float*)ws;
  __bf16* P    = (__bf16*)ws;
  __bf16* g    = (__bf16*)ws;
  float*  att  = (float*)(ws + MB(128));
  float*  f2   = att;
  float*  h    = (float*)(ws + MB(192));
  __bf16* x_bf = (__bf16*)(ws + MB(256));
  __bf16* xT   = (__bf16*)(ws + MB(288));
  __bf16* h_bf = (__bf16*)(ws + MB(320));
  __bf16* w1t  = (__bf16*)(ws + MB(352));
  __bf16* w2t  = (__bf16*)(ws + MB(360));

  // prep: bf16 packs (bandwidth-trivial vs the WMMA work)
  cvt_bf16_kernel      <<<4096, 256, 0, stream>>>(x, x_bf, (long)B_ * N_ * H_);
  transpose_cvt_kernel <<<dim3(H_ / 32, N_ / 32, B_), 256, 0, stream>>>(x, xT, N_, H_);
  transpose_cvt_kernel <<<dim3(FF_ / 32, H_ / 32, 1), 256, 0, stream>>>(w1, w1t, H_, FF_);
  transpose_cvt_kernel <<<dim3(H_ / 32, FF_ / 32, 1), 256, 0, stream>>>(w2, w2t, FF_, H_);

  scores_kernel <<<dim3(N_ / 128, N_ / 128, B_), 256, 0, stream>>>(x_bf, S);
  softmax_kernel<<<dim3(B_ * N_), 256, 0, stream>>>(S);
  attend_kernel <<<dim3(H_ / 128, N_ / 128, B_), 256, 0, stream>>>(P, xT, att);
  add_ln_kernel <<<dim3(B_ * N_), 256, 0, stream>>>(x, att, ln1_g, ln1_b, h, h_bf);
  ffn1_kernel   <<<dim3(FF_ / 128, (B_ * N_) / 128), 256, 0, stream>>>(h_bf, w1t, b1, g);
  ffn2_kernel   <<<dim3(H_ / 128, (B_ * N_) / 128), 256, 0, stream>>>(g, w2t, b2, f2);
  add_ln_kernel <<<dim3(B_ * N_), 256, 0, stream>>>(h, f2, ln2_g, ln2_b, out, (__bf16*)nullptr);
}
